// GNNLayer_66838281061309
// MI455X (gfx1250) — matrix operand order repaired and checked by reading, not compiled, not run on hardware
//
#include <hip/hip_runtime.h>
#include <stdint.h>

typedef __attribute__((ext_vector_type(16))) _Float16 v16h;
typedef __attribute__((ext_vector_type(8)))  float    v8f;
typedef __attribute__((ext_vector_type(4)))  float    v4f;

#define N_NODES 20000
#define KNB 16
#define DIM 128
#define DEE 64
#define NH 8
#define NA 16
#define FTOT 320
#define NKB 10            // FTOT / 32
#define EPSB 1e-5f

// ---- workspace layout (float offsets unless noted) ----
#define WS_PROD   0ULL                      // N*K*128 floats (prod); overlays e-buffer
#define WS_E      0ULL                      // N*K*64 floats (e), dead before prod written
#define WS_HEADS  40960000ULL               // N*128 floats
#define WS_MASK   43520000ULL               // N*K floats
#define WS_STATS  43840000ULL               // 2048 floats
#define WS_PACK_BYTES ((43840000ULL + 2048ULL) * 4ULL)   // packed f16 weights (32B aligned)
// stats sub-offsets (floats within stats)
#define ST_SE    0
#define ST_S2E   64
#define ST_SATT  128
#define ST_S2ATT 256
#define ST_SOUT  384
#define ST_S2OUT 512
#define ST_CNT   640
#define ST_AE    656
#define ST_BE    720
#define ST_AATT  784
#define ST_BATT  912
#define ST_AOUT  1040
#define ST_BOUT  1168
// packed-weight sub-offsets (halves)
#define PK_EDGE_HALF 0
#define PK_ATT_HALF  20480
#define PK_VAL_HALF  61440

__device__ __forceinline__ float sp(float x) {           // stable softplus
  return fmaxf(x, 0.f) + log1pf(__expf(-fabsf(x)));
}

__device__ __forceinline__ v8f wmma_f16(v16h a, v16h b, v8f c) {
  return __builtin_amdgcn_wmma_f32_16x16x32_f16(false, a, false, b, (short)0, c,
                                                false, false);
}

// Load 8 consecutive f32 of a total2 row: [x_i (0..127) | x_j (128..255) | edge (256..319)]
__device__ __forceinline__ void load8(const float* __restrict__ nf, int jrow,
                                      const float* __restrict__ ebuf,
                                      int n, int r, int f0, float* o) {
  const float* s;
  if (f0 < DIM)            s = nf   + (size_t)n * DIM + f0;
  else if (f0 < 2 * DIM)   s = nf   + (size_t)jrow * DIM + (f0 - DIM);
  else                     s = ebuf + (size_t)(n * KNB + r) * DEE + (f0 - 2 * DIM);
  v4f a = *(const v4f*)s;
  v4f b = *(const v4f*)(s + 4);
  o[0]=a.x; o[1]=a.y; o[2]=a.z; o[3]=a.w;
  o[4]=b.x; o[5]=b.y; o[6]=b.z; o[7]=b.w;
}

// Build a 16x32 f16 A-fragment (contiguous K order: lane=row%16, hi selects K sub-blocks)
__device__ __forceinline__ v16h build_afrag(const float* __restrict__ nf, int jrow,
                                            const float* __restrict__ ebuf,
                                            int n, int r, int hi, int kb) {
  float t[16];
  load8(nf, jrow, ebuf, n, r, kb * 32 + 16 * hi, t);          // elems 0..7  -> K = 16hi+q
  load8(nf, jrow, ebuf, n, r, kb * 32 + 16 * hi + 8, t + 8); // elems 8..15 -> K = 16hi+q
  v16h v;
#pragma unroll
  for (int q = 0; q < 16; q++) v[q] = (_Float16)t[q];
  return v;
}

// ---------------- kernel 0a: zero the statistics accumulators ----------------
__global__ __launch_bounds__(256) void k0_zero(float* __restrict__ stats) {
  for (int i = threadIdx.x; i <= ST_CNT; i += 256) stats[i] = 0.f;
}

// ---------------- kernel 0b: pack weights into WMMA B-fragment order ----------------
// B layout per fragment (32 lanes x 16 halves): N = lane%16, K = elem + 16*(lane/16)
// slot(kb,nt) = (nt>>2)*40 + kb*4 + (nt&3)  (phased for 40KB LDS staging)
__global__ __launch_bounds__(256) void k0_pack(const float* __restrict__ edge_w,
                                               const float* __restrict__ att_w1,
                                               const float* __restrict__ val_w,
                                               _Float16* __restrict__ pk) {
  int i = blockIdx.x * 256 + threadIdx.x;
  if (i < 20480) {                       // edge_w: (320,64), 4 N-tiles, single phase
    int e = i & 15, L = (i >> 4) & 31, slot = i >> 9;     // slot = kb*4 + nt
    int kb = slot >> 2, nt = slot & 3;
    int f = kb * 32 + e + 16 * (L >> 4);
    int c = nt * 16 + (L & 15);
    pk[PK_EDGE_HALF + i] = (_Float16)edge_w[f * DEE + c];
    return;
  }
  int i2 = i - 20480;
  if (i2 < 40960) {                      // att_w1: (H,F,A) -> W[f][h*16+a], 8 N-tiles
    int e = i2 & 15, L = (i2 >> 4) & 31, slot = i2 >> 9;  // 0..79
    int ph = slot / 40, rm = slot % 40;
    int kb = rm >> 2, nt = ph * 4 + (rm & 3);
    int f = kb * 32 + e + 16 * (L >> 4);
    int c = nt * 16 + (L & 15);
    int h = c >> 4, a = c & 15;
    pk[PK_ATT_HALF + i2] = (_Float16)att_w1[((size_t)h * FTOT + f) * NA + a];
    return;
  }
  int i3 = i2 - 40960;
  if (i3 < 40960) {                      // val_w
    int e = i3 & 15, L = (i3 >> 4) & 31, slot = i3 >> 9;
    int ph = slot / 40, rm = slot % 40;
    int kb = rm >> 2, nt = ph * 4 + (rm & 3);
    int f = kb * 32 + e + 16 * (L >> 4);
    int c = nt * 16 + (L & 15);
    int h = c >> 4, a = c & 15;
    pk[PK_VAL_HALF + i3] = (_Float16)val_w[((size_t)h * FTOT + f) * NA + a];
  }
}

// ---------------- kernel 1: e = total @ edge_w + b, mask, masked-BN partials ----------------
__global__ __launch_bounds__(256) void k1_edge_gemm(
    const float* __restrict__ nf, const float* __restrict__ ef,
    const float* __restrict__ nmask, const int* __restrict__ nbr,
    const float* __restrict__ edge_b, const _Float16* __restrict__ pk,
    float* __restrict__ ebuf_out, float* __restrict__ maskbuf,
    float* __restrict__ stats) {
  __shared__ __align__(32) uint32_t lB[10240];   // 40KB packed edge B-fragments
  __shared__ float lstat[130];                   // 64 sum | 64 sumsq | cnt
  {
    const uint4* g4 = (const uint4*)(pk + PK_EDGE_HALF);
    uint4* l4 = (uint4*)lB;
    for (int i = threadIdx.x; i < 2560; i += 256) l4[i] = g4[i];
  }
  if (threadIdx.x < 130) lstat[threadIdx.x] = 0.f;
  __syncthreads();

  int w = threadIdx.x >> 5, lane = threadIdx.x & 31;
  int r = lane & 15, hi = lane >> 4;
  int n = blockIdx.x * 8 + w;

  // mask: where(lens==0, first-one, neighbor_masks)
  float mk = (lane < 16) ? nmask[n * KNB + lane] : 0.f;
  float lens = mk;
#pragma unroll
  for (int m = 1; m < 32; m <<= 1) lens += __shfl_xor(lens, m, 32);
  if (lane < 16 && lens < 0.5f) mk = (lane == 0) ? 1.f : 0.f;
  if (lane < 16) maskbuf[n * KNB + lane] = mk;
  float mrow[8];
#pragma unroll
  for (int rr = 0; rr < 8; rr++) mrow[rr] = __shfl(mk, rr + 8 * hi, 32);
  float cnt_n = fmaxf(lens, 1.f);

  int j = nbr[n * KNB + r];
  v16h af[NKB];
#pragma unroll
  for (int kb = 0; kb < NKB; kb++) af[kb] = build_afrag(nf, j, ef, n, r, hi, kb);

#pragma unroll
  for (int nt = 0; nt < 4; nt++) {
    v8f acc = {};
#pragma unroll
    for (int kb = 0; kb < NKB; kb++) {
      v16h b = *(const v16h*)&lB[((kb * 4 + nt) * 32 + lane) * 8];
      acc = wmma_f16(af[kb], b, acc);
    }
    int c = nt * 16 + r;
    float bias = edge_b[c];
    float ps = 0.f, pq = 0.f;
#pragma unroll
    for (int rr = 0; rr < 8; rr++) {
      float v = acc[rr] + bias;
      ebuf_out[(size_t)(n * KNB + rr + 8 * hi) * DEE + c] = v;
      float vm = v * mrow[rr];
      ps += vm; pq += v * vm;
    }
    ps += __shfl_xor(ps, 16, 32);
    pq += __shfl_xor(pq, 16, 32);
    if (hi == 0) { atomicAdd(&lstat[c], ps); atomicAdd(&lstat[64 + c], pq); }
  }
  if (lane == 0) atomicAdd(&lstat[128], cnt_n);
  __syncthreads();
  if (threadIdx.x < 64) {
    atomicAdd(&stats[ST_SE + threadIdx.x],  lstat[threadIdx.x]);
    atomicAdd(&stats[ST_S2E + threadIdx.x], lstat[64 + threadIdx.x]);
  } else if (threadIdx.x == 64) {
    atomicAdd(&stats[ST_CNT], lstat[128]);
  }
}

// ---------------- kernel 2: finalize edge BN -> scale/bias ----------------
__global__ void k2_fin_edge(const float* __restrict__ edge_g,
                            const float* __restrict__ edge_beta,
                            float* __restrict__ stats) {
  int c = threadIdx.x;                   // 64 threads
  float cnt = fmaxf(stats[ST_CNT], 1.f);
  float mean = stats[ST_SE + c] / cnt;
  float var = fmaxf(stats[ST_S2E + c] / cnt - mean * mean, 0.f);
  float a = edge_g[c] * rsqrtf(var + EPSB);
  stats[ST_AE + c] = a;
  stats[ST_BE + c] = edge_beta[c] - mean * a;
}

// ---------------- kernel 3: edge_out = softplus(edge_f + BN(e)*mask) ----------------
__global__ __launch_bounds__(256) void k3_edge_out(
    const float* __restrict__ ebuf, const float* __restrict__ ef,
    const float* __restrict__ maskbuf, const float* __restrict__ stats,
    float* __restrict__ eout) {
  __shared__ float sa[64], sb[64];
  if (threadIdx.x < 64) {
    sa[threadIdx.x] = stats[ST_AE + threadIdx.x];
    sb[threadIdx.x] = stats[ST_BE + threadIdx.x];
  }
  __syncthreads();
  int i = blockIdx.x * 256 + threadIdx.x;          // 320000 rows
  float mk = maskbuf[i];
  const float* er = ebuf + (size_t)i * DEE;
  const float* fr = ef + (size_t)i * DEE;
  float* orow = eout + (size_t)i * DEE;
#pragma unroll
  for (int c4 = 0; c4 < 16; c4++) {
    v4f e4 = *(const v4f*)(er + c4 * 4);
    v4f f4 = *(const v4f*)(fr + c4 * 4);
    v4f o;
#pragma unroll
    for (int u = 0; u < 4; u++) {
      int c = c4 * 4 + u;
      float bn = e4[u] * sa[c] + sb[c];
      o[u] = sp(f4[u] + bn * mk);
    }
    *(v4f*)(orow + c4 * 4) = o;
  }
}

// ---------------- kernel 4: att + val GEMMs, softmax, prod, masked-BN partials ----------------
__global__ __launch_bounds__(256) void k4_att(
    const float* __restrict__ nf, const float* __restrict__ eo,
    const float* __restrict__ maskbuf, const int* __restrict__ nbr,
    const float* __restrict__ att_b1, const float* __restrict__ att_w2,
    const float* __restrict__ att_b2, const float* __restrict__ val_b,
    const _Float16* __restrict__ pk, float* __restrict__ prod,
    float* __restrict__ stats) {
  __shared__ __align__(32) uint32_t lB[10240];     // 40KB staging (one phase)
  __shared__ float latt[8][KNB][NH];               // per-node scores -> att weights
  __shared__ float lstat[256];
  lstat[threadIdx.x] = 0.f;

  int w = threadIdx.x >> 5, lane = threadIdx.x & 31;
  int r = lane & 15, hi = lane >> 4;
  int n = blockIdx.x * 8 + w;

  float mrow[8];
#pragma unroll
  for (int rr = 0; rr < 8; rr++) mrow[rr] = maskbuf[n * KNB + rr + 8 * hi];

  int j = nbr[n * KNB + r];
  v16h af[NKB];
#pragma unroll
  for (int kb = 0; kb < NKB; kb++) af[kb] = build_afrag(nf, j, eo, n, r, hi, kb);

  const uint4* gatt = (const uint4*)(pk + PK_ATT_HALF);
  const uint4* gval = (const uint4*)(pk + PK_VAL_HALF);

  // ---- attention hidden -> per-head scores ----
  for (int ph = 0; ph < 2; ph++) {
    __syncthreads();
    { uint4* l4 = (uint4*)lB;
      for (int i = threadIdx.x; i < 2560; i += 256) l4[i] = gatt[ph * 2560 + i]; }
    __syncthreads();
#pragma unroll
    for (int ntl = 0; ntl < 4; ntl++) {
      int t = ph * 4 + ntl;                        // head index
      v8f acc = {};
#pragma unroll
      for (int kb = 0; kb < NKB; kb++) {
        v16h b = *(const v16h*)&lB[((kb * 4 + ntl) * 32 + lane) * 8];
        acc = wmma_f16(af[kb], b, acc);
      }
      int c = t * 16 + r;
      float b1 = att_b1[c];
      float w2 = att_w2[c];
      float b2 = att_b2[t];
#pragma unroll
      for (int rr = 0; rr < 8; rr++) {
        float hv = sp(acc[rr] + b1) * w2;
#pragma unroll
        for (int m = 1; m < 16; m <<= 1) hv += __shfl_xor(hv, m, 32);
        if (r == 0) {
          float sc = (mrow[rr] > 0.f) ? (hv + b2) : -1e30f;
          latt[w][rr + 8 * hi][t] = sc;
        }
      }
    }
  }
  __syncthreads();
  // ---- softmax over K per (node, head) ----
  if (lane < NH) {
    float mx = -3e30f;
#pragma unroll
    for (int k = 0; k < KNB; k++) mx = fmaxf(mx, latt[w][k][lane]);
    float sm = 0.f;
#pragma unroll
    for (int k = 0; k < KNB; k++) {
      float ex = __expf(latt[w][k][lane] - mx);
      latt[w][k][lane] = ex; sm += ex;
    }
    float inv = 1.f / sm;
#pragma unroll
    for (int k = 0; k < KNB; k++) latt[w][k][lane] *= inv;
  }
  // ---- values GEMM, prod = att * vals, masked-BN partials ----
  for (int ph = 0; ph < 2; ph++) {
    __syncthreads();
    { uint4* l4 = (uint4*)lB;
      for (int i = threadIdx.x; i < 2560; i += 256) l4[i] = gval[ph * 2560 + i]; }
    __syncthreads();
#pragma unroll
    for (int ntl = 0; ntl < 4; ntl++) {
      int t = ph * 4 + ntl;
      v8f acc = {};
#pragma unroll
      for (int kb = 0; kb < NKB; kb++) {
        v16h b = *(const v16h*)&lB[((kb * 4 + ntl) * 32 + lane) * 8];
        acc = wmma_f16(af[kb], b, acc);
      }
      int c = t * 16 + r;
      float vb = val_b[c];
      float ps = 0.f, pq = 0.f;
#pragma unroll
      for (int rr = 0; rr < 8; rr++) {
        float attv = latt[w][rr + 8 * hi][t];
        float p = (acc[rr] + vb) * attv;
        prod[(size_t)(n * KNB + rr + 8 * hi) * (NH * NA) + c] = p;
        float pm = p * mrow[rr];
        ps += pm; pq += p * pm;
      }
      ps += __shfl_xor(ps, 16, 32);
      pq += __shfl_xor(pq, 16, 32);
      if (hi == 0) { atomicAdd(&lstat[c], ps); atomicAdd(&lstat[128 + c], pq); }
    }
  }
  __syncthreads();
  if (threadIdx.x < 128) atomicAdd(&stats[ST_SATT + threadIdx.x], lstat[threadIdx.x]);
  else                   atomicAdd(&stats[ST_S2ATT + threadIdx.x - 128], lstat[threadIdx.x]);
}

// ---------------- kernel 5: finalize att BN ----------------
__global__ void k5_fin_att(const float* __restrict__ att_g,
                           const float* __restrict__ att_beta,
                           float* __restrict__ stats) {
  int c = threadIdx.x;                   // 128 threads
  float cnt = fmaxf(stats[ST_CNT], 1.f);
  float mean = stats[ST_SATT + c] / cnt;
  float var = fmaxf(stats[ST_S2ATT + c] / cnt - mean * mean, 0.f);
  float a = att_g[c] * rsqrtf(var + EPSB);
  stats[ST_AATT + c] = a;
  stats[ST_BATT + c] = att_beta[c] - mean * a;
}

// ---------------- kernel 6: headf = softplus(BN(prod))*mask, sum over K; out-BN partials ----------------
__global__ __launch_bounds__(256) void k6_headsum(
    const float* __restrict__ prod, const float* __restrict__ maskbuf,
    float* __restrict__ stats, float* __restrict__ heads) {
  __shared__ float sa[128], sb[128];
  __shared__ float lstat[256];
  lstat[threadIdx.x] = 0.f;
  if (threadIdx.x < 128) {
    sa[threadIdx.x] = stats[ST_AATT + threadIdx.x];
    sb[threadIdx.x] = stats[ST_BATT + threadIdx.x];
  }
  __syncthreads();
  int w = threadIdx.x >> 5, lane = threadIdx.x & 31;
  int n = blockIdx.x * 8 + w;
  int c0 = lane * 4;
  float acc[4] = {0.f, 0.f, 0.f, 0.f};
  for (int k = 0; k < KNB; k++) {
    float mkv = maskbuf[n * KNB + k];
    v4f p = *(const v4f*)(prod + (size_t)(n * KNB + k) * 128 + c0);
#pragma unroll
    for (int u = 0; u < 4; u++)
      acc[u] += sp(p[u] * sa[c0 + u] + sb[c0 + u]) * mkv;
  }
  v4f h4;
#pragma unroll
  for (int u = 0; u < 4; u++) h4[u] = acc[u];
  *(v4f*)(heads + (size_t)n * 128 + c0) = h4;
#pragma unroll
  for (int u = 0; u < 4; u++) {
    atomicAdd(&lstat[c0 + u], acc[u]);
    atomicAdd(&lstat[128 + c0 + u], acc[u] * acc[u]);
  }
  __syncthreads();
  if (threadIdx.x < 128) atomicAdd(&stats[ST_SOUT + threadIdx.x], lstat[threadIdx.x]);
  else                   atomicAdd(&stats[ST_S2OUT + threadIdx.x - 128], lstat[threadIdx.x]);
}

// ---------------- kernel 7: finalize output BN ----------------
__global__ void k7_fin_out(const float* __restrict__ out_g,
                           const float* __restrict__ out_beta,
                           float* __restrict__ stats) {
  int c = threadIdx.x;                   // 128 threads
  float invn = 1.f / (float)N_NODES;
  float mean = stats[ST_SOUT + c] * invn;
  float var = fmaxf(stats[ST_S2OUT + c] * invn - mean * mean, 0.f);
  float a = out_g[c] * rsqrtf(var + EPSB);
  stats[ST_AOUT + c] = a;
  stats[ST_BOUT + c] = out_beta[c] - mean * a;
}

// ---------------- kernel 8: atom_out = node_features + BN(heads) ----------------
__global__ __launch_bounds__(256) void k8_atom(
    const float* __restrict__ nf, const float* __restrict__ heads,
    const float* __restrict__ stats, float* __restrict__ out) {
  __shared__ float sa[128], sb[128];
  if (threadIdx.x < 128) {
    sa[threadIdx.x] = stats[ST_AOUT + threadIdx.x];
    sb[threadIdx.x] = stats[ST_BOUT + threadIdx.x];
  }
  __syncthreads();
  int i = blockIdx.x * 256 + threadIdx.x;          // 2,560,000
  int c = i & 127;
  out[i] = nf[i] + heads[i] * sa[c] + sb[c];
}

extern "C" void kernel_launch(void* const* d_in, const int* in_sizes, int n_in,
                              void* d_out, int out_size, void* d_ws, size_t ws_size,
                              hipStream_t stream) {
  (void)in_sizes; (void)n_in; (void)out_size; (void)ws_size;
  const float* nf        = (const float*)d_in[0];
  const float* ef        = (const float*)d_in[1];
  const float* nmask     = (const float*)d_in[2];
  // d_in[3] = h, d_in[4] = c : unused by the reference math
  const float* edge_w    = (const float*)d_in[5];
  const float* edge_b    = (const float*)d_in[6];
  const float* edge_g    = (const float*)d_in[7];
  const float* edge_beta = (const float*)d_in[8];
  const float* att_w1    = (const float*)d_in[9];
  const float* att_b1    = (const float*)d_in[10];
  const float* att_w2    = (const float*)d_in[11];
  const float* att_b2    = (const float*)d_in[12];
  const float* val_w     = (const float*)d_in[13];
  const float* val_b     = (const float*)d_in[14];
  const float* att_g     = (const float*)d_in[15];
  const float* att_beta  = (const float*)d_in[16];
  const float* out_g     = (const float*)d_in[17];
  const float* out_beta  = (const float*)d_in[18];
  const int*   nbr       = (const int*)d_in[19];

  float* ws      = (float*)d_ws;
  float* prod    = ws + WS_PROD;
  float* ebuf    = ws + WS_E;        // overlays prod; dead before prod written
  float* heads   = ws + WS_HEADS;
  float* maskbuf = ws + WS_MASK;
  float* stats   = ws + WS_STATS;
  _Float16* pk   = (_Float16*)((char*)d_ws + WS_PACK_BYTES);

  float* atom_out = (float*)d_out;
  float* edge_out = atom_out + (size_t)N_NODES * DIM;

  k0_zero<<<1, 256, 0, stream>>>(stats);
  k0_pack<<<400, 256, 0, stream>>>(edge_w, att_w1, val_w, pk);
  k1_edge_gemm<<<N_NODES / 8, 256, 0, stream>>>(nf, ef, nmask, nbr, edge_b, pk,
                                                ebuf, maskbuf, stats);
  k2_fin_edge<<<1, 64, 0, stream>>>(edge_g, edge_beta, stats);
  k3_edge_out<<<(N_NODES * KNB) / 256, 256, 0, stream>>>(ebuf, ef, maskbuf, stats,
                                                         edge_out);
  k4_att<<<N_NODES / 8, 256, 0, stream>>>(nf, edge_out, maskbuf, nbr, att_b1,
                                          att_w2, att_b2, val_b, pk, prod, stats);
  k5_fin_att<<<1, 128, 0, stream>>>(att_g, att_beta, stats);
  k6_headsum<<<N_NODES / 8, 256, 0, stream>>>(prod, maskbuf, stats, heads);
  k7_fin_out<<<1, 128, 0, stream>>>(out_g, out_beta, stats);
  k8_atom<<<(N_NODES * DIM) / 256, 256, 0, stream>>>(nf, heads, stats, atom_out);
}
